// Encoder_70995809403432
// MI455X (gfx1250) — matrix operand order repaired and checked by reading, not compiled, not run on hardware
//
#include <hip/hip_runtime.h>

typedef unsigned short u16;
typedef __attribute__((ext_vector_type(16))) __bf16 v16bf;
typedef __attribute__((ext_vector_type(8)))  float  v8f;

union Frag { v16bf v; uint4 q[2]; u16 h[16]; };

static constexpr int TOK  = 4096;   // 2 batches * 2048 tokens
static constexpr int SEQ  = 2048;
static constexpr int D    = 1024;
static constexpr int INNER= 1024;
static constexpr int DFF  = 4096;
static constexpr int DH   = 64;

__device__ inline u16 f2bf(float f) {
  union { float f; unsigned u; } x; x.f = f;
  unsigned r = x.u + 0x7FFFu + ((x.u >> 16) & 1u);   // round-to-nearest-even
  return (u16)(r >> 16);
}

__device__ inline void lds_fence_wave() {
  // wave-local LDS RAW ordering (store P -> reload as A-operand)
  asm volatile("s_wait_dscnt 0x0" ::: "memory");
}

// CDNA5 async global->LDS copy (ASYNCcnt-tracked DMA path, ISA ch.10 async ops)
__device__ inline void async_load_b128(const void* gptr, void* lptr) {
  unsigned lds = (unsigned)(unsigned long long)(uintptr_t)lptr; // low 32b = LDS offset
  unsigned long long ga = (unsigned long long)(uintptr_t)gptr;
  asm volatile("global_load_async_to_lds_b128 %0, %1, off"
               :: "v"(lds), "v"(ga) : "memory");
}
__device__ inline void wait_async0() {
  asm volatile("s_wait_asynccnt 0x0" ::: "memory");
}

// ---------------------------------------------------------------------------
// LayerNorm (fp32 in) -> bf16 out.  One block per token row, 256 threads.
// ---------------------------------------------------------------------------
__global__ __launch_bounds__(256)
void ln_cast(const float* __restrict__ x, const float* __restrict__ g,
             const float* __restrict__ b, u16* __restrict__ out)
{
  __shared__ float red[8];
  const int row = blockIdx.x, t = threadIdx.x, lane = t & 31, w = t >> 5;
  float4 v = ((const float4*)(x + (size_t)row * D))[t];
  float s = v.x + v.y + v.z + v.w;
#pragma unroll
  for (int m = 1; m < 32; m <<= 1) s += __shfl_xor(s, m, 32);
  if (lane == 0) red[w] = s;
  __syncthreads();
  float mean = 0.f;
#pragma unroll
  for (int i = 0; i < 8; ++i) mean += red[i];
  mean *= (1.0f / 1024.0f);
  __syncthreads();
  float dx = v.x - mean, dy = v.y - mean, dz = v.z - mean, dw = v.w - mean;
  float q = dx*dx + dy*dy + dz*dz + dw*dw;
#pragma unroll
  for (int m = 1; m < 32; m <<= 1) q += __shfl_xor(q, m, 32);
  if (lane == 0) red[w] = q;
  __syncthreads();
  float var = 0.f;
#pragma unroll
  for (int i = 0; i < 8; ++i) var += red[i];
  var *= (1.0f / 1024.0f);
  float rstd = rsqrtf(var + 1e-5f);
  float4 gg = ((const float4*)g)[t];
  float4 bb = ((const float4*)b)[t];
  u16* op = out + (size_t)row * D + t * 4;
  op[0] = f2bf(dx * rstd * gg.x + bb.x);
  op[1] = f2bf(dy * rstd * gg.y + bb.y);
  op[2] = f2bf(dz * rstd * gg.z + bb.z);
  op[3] = f2bf(dw * rstd * gg.w + bb.w);
}

// ---------------------------------------------------------------------------
// bf16 WMMA GEMM: out[M,N] = A[M,K](bf16) @ W[K,N](f32, cast to bf16 on load)
// MODE 0: -> bf16 ; MODE 1: +bias, exact GELU -> bf16 ; MODE 2: +bias+resid -> f32
// Block tile 128x128, K-step 64, double-buffered LDS pipeline:
//   issue async A-DMA + W loads for tile k+1  ->  16 WMMAs on tile k
//   ->  convert/store next B tile  ->  s_wait_asynccnt  ->  one barrier.
// ---------------------------------------------------------------------------
template<int MODE>
__global__ __launch_bounds__(256)
void gemm_bf16(const u16* __restrict__ A, const float* __restrict__ W,
               const float* __restrict__ bias, const float* __restrict__ resid,
               void* __restrict__ outv, int M, int N, int K)
{
  constexpr int BM = 128, BN = 128, BK = 64, LDA = 72, LDB = 72;
  __shared__ u16 As[2][BM * LDA];   // [m][k], padded (2 x 18 KB)
  __shared__ u16 Bs[2][BN * LDB];   // transposed: [n][k], padded (2 x 18 KB)
  const int t = threadIdx.x, lane = t & 31, w = t >> 5;
  const int half = lane >> 4, l16 = lane & 15;
  const int wr = w >> 1, wc = w & 1;
  const int bm = blockIdx.y * BM, bn = blockIdx.x * BN;

  v8f acc[2][4] = {};
  const int arow = t >> 1, ach = (t & 1) * 32;   // A: 128 rows x 64, 32 elems/thread
  const int bk   = t >> 2, bn32 = (t & 3) * 32;  // B: 64 rows  x 128, 32 elems/thread
  const u16*    agbase = A + (size_t)(bm + arow) * K + ach;
  const float4* wgbase = (const float4*)(W + (size_t)bk * N + bn + bn32);
  const size_t  wgstep = (size_t)BK * N / 4;     // float4 step per k-tile

  // ---- prologue: fill buffer 0 with tile kt=0 ----
  {
#pragma unroll
    for (int u = 0; u < 4; ++u)
      async_load_b128(agbase + u * 8, &As[0][arow * LDA + ach + u * 8]);
#pragma unroll
    for (int v4 = 0; v4 < 8; ++v4) {
      float4 wv = wgbase[v4];
      int nb = bn32 + v4 * 4;
      Bs[0][(nb + 0) * LDB + bk] = f2bf(wv.x);
      Bs[0][(nb + 1) * LDB + bk] = f2bf(wv.y);
      Bs[0][(nb + 2) * LDB + bk] = f2bf(wv.z);
      Bs[0][(nb + 3) * LDB + bk] = f2bf(wv.w);
    }
    wait_async0();
    __syncthreads();
  }

  int cur = 0;
  for (int kt = 0; kt < K; kt += BK) {
    const int nxt = cur ^ 1;
    const bool has_next = (kt + BK) < K;
    float4 wreg[8];
    if (has_next) {
      const u16* ag = agbase + (kt + BK);
#pragma unroll
      for (int u = 0; u < 4; ++u)
        async_load_b128(ag + u * 8, &As[nxt][arow * LDA + ach + u * 8]);
      const float4* wg = wgbase + (size_t)(kt / BK + 1) * wgstep;
#pragma unroll
      for (int v4 = 0; v4 < 8; ++v4) wreg[v4] = wg[v4];
    }

    // ---- compute on buffer `cur` (2 K-chunks x 2x4 subtiles = 16 WMMAs) ----
#pragma unroll
    for (int c = 0; c < 2; ++c) {
      Frag a[2], b[4];
#pragma unroll
      for (int i = 0; i < 2; ++i) {          // A-operand layout (16-bit, 16x32)
        int mr = wr * 32 + i * 16 + l16;
        a[i].q[0] = *(const uint4*)&As[cur][mr * LDA + c * 32 + half * 8];
        a[i].q[1] = *(const uint4*)&As[cur][mr * LDA + c * 32 + half * 8 + 16];
      }
#pragma unroll
      for (int j = 0; j < 4; ++j) {          // B-operand layout (32x16)
        int nc = wc * 64 + j * 16 + l16;
        b[j].q[0] = *(const uint4*)&Bs[cur][nc * LDB + c * 32 + half * 8];
        b[j].q[1] = *(const uint4*)&Bs[cur][nc * LDB + c * 32 + half * 8 + 16];
      }
#pragma unroll
      for (int i = 0; i < 2; ++i)
#pragma unroll
        for (int j = 0; j < 4; ++j)
          acc[i][j] = __builtin_amdgcn_wmma_f32_16x16x32_bf16(
              false, a[i].v, false, b[j].v, (short)0, acc[i][j], false, false);
    }

    if (has_next) {
#pragma unroll
      for (int v4 = 0; v4 < 8; ++v4) {
        int nb = bn32 + v4 * 4;
        Bs[nxt][(nb + 0) * LDB + bk] = f2bf(wreg[v4].x);
        Bs[nxt][(nb + 1) * LDB + bk] = f2bf(wreg[v4].y);
        Bs[nxt][(nb + 2) * LDB + bk] = f2bf(wreg[v4].z);
        Bs[nxt][(nb + 3) * LDB + bk] = f2bf(wreg[v4].w);
      }
      wait_async0();
    }
    __syncthreads();          // retires reads of `cur`, publishes writes to `nxt`
    cur = nxt;
  }

#pragma unroll
  for (int i = 0; i < 2; ++i)
#pragma unroll
    for (int j = 0; j < 4; ++j) {
      int n = bn + wc * 64 + j * 16 + l16;
#pragma unroll
      for (int r = 0; r < 8; ++r) {
        int m = bm + wr * 32 + i * 16 + half * 8 + r;   // C/D layout
        float val = acc[i][j][r];
        if (MODE == 1) {
          val += bias[n];
          val = 0.5f * val * (1.0f + erff(val * 0.70710678118654752f)); // exact GELU
        }
        if (MODE == 2) {
          val += bias[n] + resid[(size_t)m * N + n];
          ((float*)outv)[(size_t)m * N + n] = val;
        } else {
          ((u16*)outv)[(size_t)m * N + n] = f2bf(val);
        }
      }
    }
}

// ---------------------------------------------------------------------------
// Flash attention: softmax(Q Kt / 8) V per (batch, head).
// Block = 8 waves x 16 query rows = 128 queries; 64-key tiles,
// double-buffered K/V pipeline (async K-DMA + V register prefetch).
// ---------------------------------------------------------------------------
__global__ __launch_bounds__(256)
void attn_kernel(const u16* __restrict__ Q, const u16* __restrict__ KV,
                 u16* __restrict__ Out)
{
  constexpr int LDK = 72;                 // padded halfword stride
  __shared__ u16 Ks[2][64 * LDK];         // [key][dh]   (== Kt B-operand storage)
  __shared__ u16 Vs[2][64 * LDK];         // [dh][key]   (transposed)
  __shared__ u16 Ps[8 * 16 * LDK];        // per-wave P scratch [16][64]
  const int t = threadIdx.x, lane = t & 31, w = t >> 5;
  const int half = lane >> 4, l16 = lane & 15;
  const int bid = blockIdx.x;
  const int qc = bid & 15, head = (bid >> 4) & 15, batch = bid >> 8;
  const int qbase = batch * SEQ + qc * 128 + w * 16;

  Frag qa[2];                             // Q[16,64] in A-layout, two 32-k chunks
  {
    const u16* qp = Q + (size_t)(qbase + l16) * INNER + head * DH;
#pragma unroll
    for (int c = 0; c < 2; ++c) {
      qa[c].q[0] = *(const uint4*)(qp + c * 32 + half * 8);
      qa[c].q[1] = *(const uint4*)(qp + c * 32 + half * 8 + 16);
    }
  }

  float m_run[8], l_run[8];
#pragma unroll
  for (int r = 0; r < 8; ++r) { m_run[r] = -3.0e38f; l_run[r] = 0.f; }
  v8f o[4] = {};
  const int key = t >> 2, seg = (t & 3) * 16;
  const float scale = 0.125f;             // 1/sqrt(64)
  const u16* kvbase = KV + (size_t)(batch * SEQ + key) * 2048 + head * DH + seg;

  // ---- prologue: fill buffer 0 with key tile 0 ----
  {
    async_load_b128(kvbase,     &Ks[0][key * LDK + seg]);
    async_load_b128(kvbase + 8, &Ks[0][key * LDK + seg + 8]);
    Frag vt;
    vt.q[0] = *(const uint4*)(kvbase + 1024);
    vt.q[1] = *(const uint4*)(kvbase + 1024 + 8);
#pragma unroll
    for (int u = 0; u < 16; ++u) Vs[0][(seg + u) * LDK + key] = vt.h[u];
    wait_async0();
    __syncthreads();
  }

  int cur = 0;
  for (int kt = 0; kt < SEQ; kt += 64) {
    const int nxt = cur ^ 1;
    const bool has_next = (kt + 64) < SEQ;
    Frag vt;
    if (has_next) {
      const u16* kp = kvbase + (size_t)(kt + 64) * 2048;
      async_load_b128(kp,     &Ks[nxt][key * LDK + seg]);
      async_load_b128(kp + 8, &Ks[nxt][key * LDK + seg + 8]);
      vt.q[0] = *(const uint4*)(kp + 1024);
      vt.q[1] = *(const uint4*)(kp + 1024 + 8);
    }

    // S[16, 64] = Q Kt  (4 key subtiles x 2 chained WMMAs)
    v8f s[4];
#pragma unroll
    for (int j = 0; j < 4; ++j) {
      v8f z = {};
#pragma unroll
      for (int c = 0; c < 2; ++c) {
        Frag bk;
        const u16* kb = &Ks[cur][(j * 16 + l16) * LDK + c * 32 + half * 8];
        bk.q[0] = *(const uint4*)kb; bk.q[1] = *(const uint4*)(kb + 16);
        z = __builtin_amdgcn_wmma_f32_16x16x32_bf16(false, qa[c].v, false, bk.v,
                                                    (short)0, z, false, false);
      }
      s[j] = z;
    }

    // online softmax: D-layout stripes keys across 16-lane halves
#pragma unroll
    for (int r = 0; r < 8; ++r) {
      float mx = m_run[r];
#pragma unroll
      for (int j = 0; j < 4; ++j) { s[j][r] *= scale; mx = fmaxf(mx, s[j][r]); }
#pragma unroll
      for (int msk = 1; msk < 16; msk <<= 1) mx = fmaxf(mx, __shfl_xor(mx, msk, 32));
      float alpha = __expf(m_run[r] - mx);
      m_run[r] = mx;
      float rs = 0.f;
#pragma unroll
      for (int j = 0; j < 4; ++j) { float p = __expf(s[j][r] - mx); s[j][r] = p; rs += p; }
#pragma unroll
      for (int msk = 1; msk < 16; msk <<= 1) rs += __shfl_xor(rs, msk, 32);
      l_run[r] = l_run[r] * alpha + rs;
#pragma unroll
      for (int jo = 0; jo < 4; ++jo) o[jo][r] *= alpha;
    }

    // D-layout -> A-layout reshape of P via per-wave LDS scratch
    u16* pw = &Ps[w * 16 * LDK];
#pragma unroll
    for (int j = 0; j < 4; ++j)
#pragma unroll
      for (int r = 0; r < 8; ++r)
        pw[(half * 8 + r) * LDK + j * 16 + l16] = f2bf(s[j][r]);
    lds_fence_wave();

    Frag pa[2];
#pragma unroll
    for (int c = 0; c < 2; ++c) {
      const u16* pp = &pw[l16 * LDK + c * 32 + half * 8];
      pa[c].q[0] = *(const uint4*)pp; pa[c].q[1] = *(const uint4*)(pp + 16);
    }
    // O[16, 64] += P V
#pragma unroll
    for (int jo = 0; jo < 4; ++jo)
#pragma unroll
      for (int c = 0; c < 2; ++c) {
        Frag bv;
        const u16* vb = &Vs[cur][(jo * 16 + l16) * LDK + c * 32 + half * 8];
        bv.q[0] = *(const uint4*)vb; bv.q[1] = *(const uint4*)(vb + 16);
        o[jo] = __builtin_amdgcn_wmma_f32_16x16x32_bf16(false, pa[c].v, false, bv.v,
                                                        (short)0, o[jo], false, false);
      }

    if (has_next) {
#pragma unroll
      for (int u = 0; u < 16; ++u) Vs[nxt][(seg + u) * LDK + key] = vt.h[u];
      wait_async0();
    }
    __syncthreads();          // retires reads of `cur`, publishes `nxt`
    cur = nxt;
  }

#pragma unroll
  for (int r = 0; r < 8; ++r) {
    float inv = 1.0f / l_run[r];
    int tok = qbase + half * 8 + r;
    u16* op = Out + (size_t)tok * INNER + head * DH;
#pragma unroll
    for (int jo = 0; jo < 4; ++jo)
      op[jo * 16 + l16] = f2bf(o[jo][r] * inv);
  }
}

// ---------------------------------------------------------------------------
extern "C" void kernel_launch(void* const* d_in, const int* in_sizes, int n_in,
                              void* d_out, int out_size, void* d_ws, size_t ws_size,
                              hipStream_t stream)
{
  const float* x    = (const float*)d_in[0];
  const float* Wq   = (const float*)d_in[1];
  const float* Wkv  = (const float*)d_in[2];
  const float* Wo   = (const float*)d_in[3];
  const float* bo   = (const float*)d_in[4];
  const float* ln1g = (const float*)d_in[5];
  const float* ln1b = (const float*)d_in[6];
  const float* ln2g = (const float*)d_in[7];
  const float* ln2b = (const float*)d_in[8];
  const float* W1   = (const float*)d_in[9];
  const float* b1   = (const float*)d_in[10];
  const float* W2   = (const float*)d_in[11];
  const float* b2   = (const float*)d_in[12];
  (void)in_sizes; (void)n_in; (void)out_size; (void)ws_size;

  float* xbuf = (float*)d_out;            // residual stream fp32, updated in place
  char* ws = (char*)d_ws;
  u16* h_bf  = (u16*)(ws);                              //  8 MB [TOK, 1024]
  u16* q_bf  = (u16*)(ws + (size_t) 8 * 1024 * 1024);   //  8 MB [TOK, 1024]
  u16* kv_bf = (u16*)(ws + (size_t)16 * 1024 * 1024);   // 16 MB [TOK, 2048]
  u16* at_bf = (u16*)(ws + (size_t)32 * 1024 * 1024);   //  8 MB [TOK, 1024]
  u16* ff_bf = (u16*)(ws + (size_t)40 * 1024 * 1024);   // 32 MB [TOK, 4096]

  hipMemcpyAsync(xbuf, x, (size_t)TOK * D * sizeof(float),
                 hipMemcpyDeviceToDevice, stream);

  for (int l = 0; l < 6; ++l) {
    ln_cast<<<TOK, 256, 0, stream>>>(xbuf, ln1g + l * D, ln1b + l * D, h_bf);
    gemm_bf16<0><<<dim3(INNER / 128, TOK / 128), 256, 0, stream>>>(
        h_bf, Wq + (size_t)l * D * INNER, nullptr, nullptr, q_bf, TOK, INNER, D);
    gemm_bf16<0><<<dim3(2 * INNER / 128, TOK / 128), 256, 0, stream>>>(
        h_bf, Wkv + (size_t)l * D * 2 * INNER, nullptr, nullptr, kv_bf, TOK, 2 * INNER, D);
    attn_kernel<<<dim3(512), 256, 0, stream>>>(q_bf, kv_bf, at_bf);
    gemm_bf16<2><<<dim3(D / 128, TOK / 128), 256, 0, stream>>>(
        at_bf, Wo + (size_t)l * INNER * D, bo + l * D, xbuf, xbuf, TOK, D, INNER);
    ln_cast<<<TOK, 256, 0, stream>>>(xbuf, ln2g + l * D, ln2b + l * D, h_bf);
    gemm_bf16<1><<<dim3(DFF / 128, TOK / 128), 256, 0, stream>>>(
        h_bf, W1 + (size_t)l * D * DFF, b1 + l * DFF, nullptr, ff_bf, TOK, DFF, D);
    gemm_bf16<2><<<dim3(D / 128, TOK / 128), 256, 0, stream>>>(
        ff_bf, W2 + (size_t)l * DFF * D, b2 + l * D, xbuf, xbuf, TOK, D, DFF);
  }
}
